// TransDecoder_59115929862519
// MI455X (gfx1250) — matrix operand order repaired and checked by reading, not compiled, not run on hardware
//
#include <hip/hip_runtime.h>
#include <hip/hip_bf16.h>

typedef __attribute__((ext_vector_type(16))) _Float16 v16h;
typedef __attribute__((ext_vector_type(8)))  float    v8f;

#define BZ   4
#define NN   128
#define DD   512
#define RELS 64
#define PP   1600     // 3*D + REL
#define KTOT 576      // 512 (span/product) + 64 (rel)
#define KCH  18       // 18 chunks of 32

__global__ __launch_bounds__(256) void trans_decoder_fused(
    const float* __restrict__ span,    // [B,N,D]
    const float* __restrict__ topic,   // [B,D]
    const int*   __restrict__ adu_len, // [B]
    const int*   __restrict__ rel,     // [B,N,N,REL]
    const float* __restrict__ W_type,  // [D]
    const float* __restrict__ b_type,  // [1]
    const float* __restrict__ W1,      // [64,1600]
    const float* __restrict__ b1,      // [64]
    const float* __restrict__ W2,      // [2,64]
    const float* __restrict__ b2,      // [2]
    float* __restrict__ out_ac,        // [B,N]
    float* __restrict__ out_pair,      // [B,N,N+1] (masked)
    float* __restrict__ out_rel)       // [B,N,N+1]
{
    __shared__ float     s_si[DD];
    __shared__ float     s_topic[DD];
    __shared__ _Float16  s_beffT[64 * KTOT];      // [k][d] : Beff transposed (72KB)
    __shared__ _Float16  s_atile[2][NN * 32];     // double-buffered A chunk (16KB)
    __shared__ float     s_csh[64];               // span_i·W1b + b1
    __shared__ float     s_w20[64];
    __shared__ float     s_w21[64];
    __shared__ float     s_hroot[64];
    __shared__ float     s_red[64];

    const int t    = threadIdx.x;
    const int wg   = blockIdx.x;
    const int b    = wg >> 7;
    const int i    = wg & 127;
    const int lane = t & 31;
    const int wv   = t >> 5;          // wave 0..7 -> j strip [16*wv, 16*wv+16)
    const int m    = lane & 15;
    const int kh   = lane >> 4;       // which 16-lane half

    // ---------- Phase A: stage span_i and topic ----------
    for (int d = t; d < DD; d += 256) {
        s_si[d]    = span[(b * NN + i) * DD + d];
        s_topic[d] = topic[b * DD + d];
    }
    __syncthreads();

    // ---------- Phase B: build Beff^T, per-i vectors, root/ac partials ----------
    // Beff[d,k] = W1a[k,d] + si[d]*W1c[k,d]  (d<512);  = W1d[k,d-512] (d>=512)
    for (int idx = t; idx < 64 * KTOT; idx += 256) {
        int k = idx / KTOT;
        int d = idx - k * KTOT;
        float v;
        if (d < DD) v = W1[k * PP + d] + s_si[d] * W1[k * PP + 1024 + d];
        else        v = W1[k * PP + 1536 + (d - DD)];
        s_beffT[idx] = (_Float16)v;
    }
    if (t < 64) {
        const int k = t;
        float s = b1[k];            // span_i·W1b + b1
        float r = b1[k];            // root pre-activation
        for (int d = 0; d < DD; ++d) {
            float si = s_si[d], tp = s_topic[d];
            s += si * W1[k * PP + 512 + d];
            r += si * W1[k * PP + d] + tp * W1[k * PP + 512 + d]
               + si * tp * W1[k * PP + 1024 + d];
        }
        s_csh[k]   = s;
        s_hroot[k] = r > 0.f ? r : 0.f;
        s_w20[k]   = W2[k];
        s_w21[k]   = W2[64 + k];
        float a = 0.f;              // ac_types partial
        for (int d = k * 8; d < k * 8 + 8; ++d) a += s_si[d] * W_type[d];
        s_red[k] = a;
    }
    __syncthreads();

    // ---------- cooperative A-chunk loader: 128 x 32 (f16), thread t -> row t/2, half-row t&1
    auto loadAtile = [&](int c, int buf) {
        const int j  = t >> 1;
        const int cb = (t & 1) * 16;
        _Float16* dst = &s_atile[buf][j * 32 + cb];
        if (c < 16) {
            const float* src = &span[(b * NN + j) * DD + c * 32 + cb];
            #pragma unroll
            for (int e = 0; e < 16; ++e) dst[e] = (_Float16)src[e];
        } else {
            const int* src = &rel[(((b * NN + i) * NN + j) * RELS) + (c - 16) * 32 + cb];
            #pragma unroll
            for (int e = 0; e < 16; ++e) dst[e] = (_Float16)src[e];
        }
    };

    // ---------- Phase C: finalize root + ac_types, prime pipeline ----------
    loadAtile(0, 0);
    if (t == 0) {
        float s0 = b2[0], s1 = b2[1];
        for (int k = 0; k < 64; ++k) { s0 += s_w20[k] * s_hroot[k]; s1 += s_w21[k] * s_hroot[k]; }
        const int row = (b * NN + i) * (NN + 1) + NN;
        out_rel[row]  = s0;
        out_pair[row] = s1;                 // root column always kept
        float ac = b_type[0];
        for (int k = 0; k < 64; ++k) ac += s_red[k];
        out_ac[b * NN + i] = ac;
    }
    __syncthreads();

    // ---------- Phase D: main WMMA loop over K (18 chunks of 32) ----------
    v8f acc[4] = {};                        // 16(j) x 64(k) per wave
    const int jrow = wv * 16 + m;           // A-matrix row this lane supplies

    for (int c = 0; c < KCH; ++c) {
        if (c + 1 < KCH) loadAtile(c + 1, (c + 1) & 1);

        // A fragment (16x32 f16): lanes 0-15 hold K 0..15, lanes 16-31 hold 16..31
        v16h af;
        const _Float16* ap = &s_atile[c & 1][jrow * 32 + kh * 16];
        #pragma unroll
        for (int e = 0; e < 16; ++e) { af[e] = ap[e]; }

        #pragma unroll
        for (int kt = 0; kt < 4; ++kt) {
            // B fragment (32x16 f16): lane n col kt*16+n; lanes 0-15 K 0..15, 16-31 K 16..31
            v16h bf;
            const _Float16* bp = &s_beffT[(kt * 16 + m) * KTOT + c * 32 + kh * 16];
            #pragma unroll
            for (int e = 0; e < 16; ++e) bf[e] = bp[e];
            acc[kt] = __builtin_amdgcn_wmma_f32_16x16x32_f16(
                false, af, false, bf, (short)0, acc[kt], false, false);
        }
        __syncthreads();
    }

    // ---------- Phase E: epilogue — ReLU, 64->2 head, mask, store ----------
    const int alen = adu_len[b];
    #pragma unroll
    for (int v = 0; v < 8; ++v) {
        float s0 = 0.f, s1 = 0.f;
        #pragma unroll
        for (int kt = 0; kt < 4; ++kt) {
            const int k = kt * 16 + m;
            float h = acc[kt][v] + s_csh[k];
            h = h > 0.f ? h : 0.f;
            s0 += s_w20[k] * h;
            s1 += s_w21[k] * h;
        }
        #pragma unroll
        for (int off = 1; off < 16; off <<= 1) {   // row-sum across the 16-lane half
            s0 += __shfl_xor(s0, off, 32);
            s1 += __shfl_xor(s1, off, 32);
        }
        if (m == 0) {
            const int j   = wv * 16 + v + 8 * kh;  // D-matrix row M = v + 8*(lane/16)
            const int row = (b * NN + i) * (NN + 1) + j;
            out_rel[row] = s0 + b2[0];
            const bool keep = (j != i) && (j < alen);
            out_pair[row] = keep ? (s1 + b2[1]) : -1e16f;
        }
    }
}

extern "C" void kernel_launch(void* const* d_in, const int* in_sizes, int n_in,
                              void* d_out, int out_size, void* d_ws, size_t ws_size,
                              hipStream_t stream) {
    const float* span   = (const float*)d_in[0];
    const float* topic  = (const float*)d_in[1];
    const int*   adu    = (const int*)  d_in[2];
    const int*   rel    = (const int*)  d_in[3];
    const float* W_type = (const float*)d_in[4];
    const float* b_type = (const float*)d_in[5];
    const float* W1     = (const float*)d_in[6];
    const float* b1     = (const float*)d_in[7];
    const float* W2     = (const float*)d_in[8];
    const float* b2     = (const float*)d_in[9];

    float* out      = (float*)d_out;
    float* out_ac   = out;                       // [4,128]
    float* out_pair = out + BZ * NN;             // [4,128,129]
    float* out_rel  = out + BZ * NN + BZ * NN * (NN + 1);

    trans_decoder_fused<<<dim3(BZ * NN), dim3(256), 0, stream>>>(
        span, topic, adu, rel, W_type, b_type, W1, b1, W2, b2,
        out_ac, out_pair, out_rel);
}
